// MultiHeadAttention_8864812498953
// MI455X (gfx1250) — matrix operand order repaired and checked by reading, not compiled, not run on hardware
//
#include <hip/hip_runtime.h>
#include <hip/hip_bf16.h>

typedef __attribute__((ext_vector_type(16))) __bf16         v16bf;
typedef __attribute__((ext_vector_type(8)))  float          v8f;
typedef __attribute__((ext_vector_type(4)))  unsigned int   v4u;
typedef __attribute__((ext_vector_type(4)))  unsigned short v4us;

#define D_MODEL  2048
#define N_HEADS  16
#define HEAD_DIM 128
#define SEQ_T    2048
#define BATCH    4
#define BT       (BATCH * SEQ_T)   // 8192 tokens

__device__ __forceinline__ unsigned short f2bf(float f) {
  unsigned int u = __float_as_uint(f);
  unsigned int lsb = (u >> 16) & 1u;
  u += 0x7fffu + lsb;               // round-to-nearest-even
  return (unsigned short)(u >> 16);
}

union FragAB { v16bf v; v4u u[2]; };

// ---- 16-lane XOR butterflies on VALU (v_permlane16_b32), not LDS -------------
#if __has_builtin(__builtin_amdgcn_permlane16)
__device__ __forceinline__ float pl16(float v, unsigned s0, unsigned s1) {
  int x = __float_as_int(v);
  return __int_as_float(__builtin_amdgcn_permlane16(x, x, s0, s1, true, false));
}
__device__ __forceinline__ float red_max16(float v) {
  v = fmaxf(v, pl16(v, 0x67452301u, 0xEFCDAB89u));  // xor 1
  v = fmaxf(v, pl16(v, 0x54761032u, 0xDCFE98BAu));  // xor 2
  v = fmaxf(v, pl16(v, 0x32107654u, 0xBA98FEDCu));  // xor 4
  v = fmaxf(v, pl16(v, 0xFEDCBA98u, 0x76543210u));  // xor 8
  return v;
}
__device__ __forceinline__ float red_sum16(float v) {
  v += pl16(v, 0x67452301u, 0xEFCDAB89u);
  v += pl16(v, 0x54761032u, 0xDCFE98BAu);
  v += pl16(v, 0x32107654u, 0xBA98FEDCu);
  v += pl16(v, 0xFEDCBA98u, 0x76543210u);
  return v;
}
#else
__device__ __forceinline__ float red_max16(float v) {
  v = fmaxf(v, __shfl_xor(v, 1, 32));
  v = fmaxf(v, __shfl_xor(v, 2, 32));
  v = fmaxf(v, __shfl_xor(v, 4, 32));
  v = fmaxf(v, __shfl_xor(v, 8, 32));
  return v;
}
__device__ __forceinline__ float red_sum16(float v) {
  v += __shfl_xor(v, 1, 32);
  v += __shfl_xor(v, 2, 32);
  v += __shfl_xor(v, 4, 32);
  v += __shfl_xor(v, 8, 32);
  return v;
}
#endif

// ---------------------------------------------------------------- cast f32->bf16 (x4)
__global__ void cast_f32_to_bf16_v4(const float* __restrict__ src,
                                    unsigned short* __restrict__ dst, int n4) {
  int i = blockIdx.x * blockDim.x + threadIdx.x;
  if (i < n4) {
    float4 f = ((const float4*)src)[i];
    v4us o = { f2bf(f.x), f2bf(f.y), f2bf(f.z), f2bf(f.w) };
    ((v4us*)dst)[i] = o;
  }
}

// ---------------------------------------------------------------- bf16 WMMA GEMM
// C[M,N] = A[M,K] * Bt[N,K]^T   (both operands K-contiguous, nn.Linear layout)
// MODE 0: C bf16 row-major; MODE 1: C bf16 scattered to (B,H,Dh,T) [V]; MODE 2: C f32 + bias
template <int MODE>
__global__ __launch_bounds__(256)
void gemm_bf16_wmma(const unsigned short* __restrict__ A,
                    const unsigned short* __restrict__ Bt,
                    void* __restrict__ Cout,
                    const float* __restrict__ bias,
                    int M, int N, int K) {
  constexpr int BM = 128, BN = 128, BK = 64, LDT = BK + 8;  // stride 72 el = 144B: 16B aligned, conflict-free
  __shared__ __align__(16) unsigned short As[2][BM][LDT];
  __shared__ __align__(16) unsigned short Bs[2][BN][LDT];

  const int t    = threadIdx.x;
  const int lane = t & 31;
  const int w    = t >> 5;
  const int lm   = lane & 15;
  const int kh   = lane >> 4;
  const int wm   = (w >> 2) * 64;
  const int wn   = (w & 3)  * 32;
  const int m0   = blockIdx.y * BM;
  const int n0   = blockIdx.x * BN;

  // tile = 128 rows x 128B -> 1024 x 16B chunks, 4 per thread
  int crow[4], coff[4];
#pragma unroll
  for (int i = 0; i < 4; ++i) { int c = t + i * 256; crow[i] = c >> 3; coff[i] = (c & 7) * 8; }

  v8f acc[4][2];
#pragma unroll
  for (int tm = 0; tm < 4; ++tm)
#pragma unroll
    for (int tn = 0; tn < 2; ++tn) { v8f z = {}; acc[tm][tn] = z; }

  // prologue: stage k-tile 0 into buffer 0
#pragma unroll
  for (int i = 0; i < 4; ++i) {
    *(v4u*)&As[0][crow[i]][coff[i]] = *(const v4u*)&A [(size_t)(m0 + crow[i]) * K + coff[i]];
    *(v4u*)&Bs[0][crow[i]][coff[i]] = *(const v4u*)&Bt[(size_t)(n0 + crow[i]) * K + coff[i]];
  }
  __syncthreads();

  const int nsteps = K / BK;
  int buf = 0;
  for (int step = 0; step < nsteps; ++step) {
    const bool haveNext = (step + 1 < nsteps);
    v4u ra[4], rb[4];
    if (haveNext) {  // issue next tile's global loads before compute (latency hiding)
      const int k0 = (step + 1) * BK;
#pragma unroll
      for (int i = 0; i < 4; ++i) {
        ra[i] = *(const v4u*)&A [(size_t)(m0 + crow[i]) * K + k0 + coff[i]];
        rb[i] = *(const v4u*)&Bt[(size_t)(n0 + crow[i]) * K + k0 + coff[i]];
      }
    }
    // 2 WMMA k-steps of 32 per staged 64-K tile
#pragma unroll
    for (int kk = 0; kk < 2; ++kk) {
      FragAB af[4], bfr[2];
#pragma unroll
      for (int tm = 0; tm < 4; ++tm) {
        af[tm].u[0] = *(const v4u*)&As[buf][wm + tm * 16 + lm][kk * 32 + kh * 8];
        af[tm].u[1] = *(const v4u*)&As[buf][wm + tm * 16 + lm][kk * 32 + 16 + kh * 8];
      }
#pragma unroll
      for (int tn = 0; tn < 2; ++tn) {
        bfr[tn].u[0] = *(const v4u*)&Bs[buf][wn + tn * 16 + lm][kk * 32 + kh * 8];
        bfr[tn].u[1] = *(const v4u*)&Bs[buf][wn + tn * 16 + lm][kk * 32 + 16 + kh * 8];
      }
#pragma unroll
      for (int tm = 0; tm < 4; ++tm)
#pragma unroll
        for (int tn = 0; tn < 2; ++tn)
          acc[tm][tn] = __builtin_amdgcn_wmma_f32_16x16x32_bf16(
              false, af[tm].v, false, bfr[tn].v, (short)0, acc[tm][tn], false, false);
    }
    if (haveNext) {  // fill the other buffer, single barrier per step
#pragma unroll
      for (int i = 0; i < 4; ++i) {
        *(v4u*)&As[buf ^ 1][crow[i]][coff[i]] = ra[i];
        *(v4u*)&Bs[buf ^ 1][crow[i]][coff[i]] = rb[i];
      }
      __syncthreads();
      buf ^= 1;
    }
  }

  // store: element (m = r + 8*kh, n = lm) of each 16x16 tile
#pragma unroll
  for (int tm = 0; tm < 4; ++tm)
#pragma unroll
    for (int tn = 0; tn < 2; ++tn)
#pragma unroll
      for (int r = 0; r < 8; ++r) {
        int   row = m0 + wm + tm * 16 + r + 8 * kh;
        int   col = n0 + wn + tn * 16 + lm;
        float val = acc[tm][tn][r];
        if constexpr (MODE == 0) {
          ((unsigned short*)Cout)[(size_t)row * N + col] = f2bf(val);
        } else if constexpr (MODE == 2) {
          ((float*)Cout)[(size_t)row * N + col] = val + bias[col];
        } else {  // MODE 1: scatter V into (B,H,Dh,T)
          int b  = row >> 11;              // / SEQ_T
          int tt = row & (SEQ_T - 1);
          int h  = col >> 7;               // / HEAD_DIM
          int dd = col & (HEAD_DIM - 1);
          ((unsigned short*)Cout)[(((size_t)b * N_HEADS + h) * HEAD_DIM + dd) * SEQ_T + tt] =
              f2bf(val);
        }
      }
}

// ---------------------------------------------------------------- flash attention
// Q,K: bf16 (B,T,D); Vt: bf16 (B,H,Dh,T); Ao: bf16 (B,T,D). 64 keys per block step.
__global__ __launch_bounds__(256)
void flash_attn_bf16(const unsigned short* __restrict__ Q,
                     const unsigned short* __restrict__ Kb,
                     const unsigned short* __restrict__ Vt,
                     unsigned short* __restrict__ Ao) {
  __shared__ __align__(16) unsigned short Ks[64][136];    // 64 keys x 128 Dh   (17.0 KB)
  __shared__ __align__(16) unsigned short Vs[128][72];    // 128 Dh x 64 keys   (18.0 KB)
  __shared__ __align__(16) unsigned short Pw[8][16][72];  // per-wave P (16x64) (18.0 KB)

  const int t    = threadIdx.x;
  const int lane = t & 31;
  const int w    = t >> 5;
  const int lm   = lane & 15;
  const int kh   = lane >> 4;

  const int bh   = blockIdx.x;
  const int b    = bh >> 4;                    // / N_HEADS
  const int h    = bh & (N_HEADS - 1);
  const int qrow = blockIdx.y * 128 + w * 16;  // wave's first Q row (in T)

  // resident Q fragments: 16 rows x 128 (Dh) = 4 WMMA k-steps
  FragAB qf[4];
  const size_t qbase = ((size_t)(b * SEQ_T + qrow + lm)) * D_MODEL + h * HEAD_DIM;
#pragma unroll
  for (int kk = 0; kk < 4; ++kk) {
    qf[kk].u[0] = *(const v4u*)&Q[qbase + kk * 32 + kh * 8];
    qf[kk].u[1] = *(const v4u*)&Q[qbase + kk * 32 + 16 + kh * 8];
  }

  float mrow[8], lrow[8];
  v8f   O[8];
#pragma unroll
  for (int r = 0; r < 8; ++r) { mrow[r] = -3.0e38f; lrow[r] = 0.0f; }
#pragma unroll
  for (int d = 0; d < 8; ++d) { v8f z = {}; O[d] = z; }

  const float scale = 0.08838834764831845f;  // 1/sqrt(128)

  for (int j = 0; j < SEQ_T / 64; ++j) {
    // stage K block: 64 rows x 256B (4x16B per thread)
#pragma unroll
    for (int i = 0; i < 4; ++i) {
      int c = t + i * 256, row = c >> 4, off = (c & 15) * 8;
      *(v4u*)&Ks[row][off] =
          *(const v4u*)&Kb[((size_t)(b * SEQ_T + j * 64 + row)) * D_MODEL + h * HEAD_DIM + off];
    }
    // stage Vt block: 128 rows x 128B (4x16B per thread)
#pragma unroll
    for (int i = 0; i < 4; ++i) {
      int c = t + i * 256, row = c >> 3, off = (c & 7) * 8;
      *(v4u*)&Vs[row][off] =
          *(const v4u*)&Vt[(((size_t)b * N_HEADS + h) * HEAD_DIM + row) * SEQ_T + j * 64 + off];
    }
    __syncthreads();

    // S = Q * K^T : 4 n-tiles of 16 keys, 16 WMMAs
    v8f s[4];
#pragma unroll
    for (int nt = 0; nt < 4; ++nt) { v8f z = {}; s[nt] = z; }
#pragma unroll
    for (int kk = 0; kk < 4; ++kk)
#pragma unroll
      for (int nt = 0; nt < 4; ++nt) {
        FragAB kf;
        kf.u[0] = *(const v4u*)&Ks[nt * 16 + lm][kk * 32 + kh * 8];
        kf.u[1] = *(const v4u*)&Ks[nt * 16 + lm][kk * 32 + 16 + kh * 8];
        s[nt] = __builtin_amdgcn_wmma_f32_16x16x32_bf16(
            false, qf[kk].v, false, kf.v, (short)0, s[nt], false, false);
      }

    // online softmax per row (row m = r + 8*kh lives within one 16-lane half)
    float al[8];
#pragma unroll
    for (int r = 0; r < 8; ++r) {
      float e0 = s[0][r] * scale, e1 = s[1][r] * scale;
      float e2 = s[2][r] * scale, e3 = s[3][r] * scale;
      float rm = red_max16(fmaxf(fmaxf(e0, e1), fmaxf(e2, e3)));
      float mnew = fmaxf(mrow[r], rm);
      float a    = __expf(mrow[r] - mnew);
      mrow[r]    = mnew;
      float p0 = __expf(e0 - mnew), p1 = __expf(e1 - mnew);
      float p2 = __expf(e2 - mnew), p3 = __expf(e3 - mnew);
      float rs = red_sum16((p0 + p1) + (p2 + p3));
      lrow[r] = lrow[r] * a + rs;
      al[r]   = a;
      Pw[w][r + 8 * kh][lm]      = f2bf(p0);
      Pw[w][r + 8 * kh][16 + lm] = f2bf(p1);
      Pw[w][r + 8 * kh][32 + lm] = f2bf(p2);
      Pw[w][r + 8 * kh][48 + lm] = f2bf(p3);
    }

    // P (C-layout) -> two A-layout fragments via wave-private LDS round trip
    FragAB pf0, pf1;
    pf0.u[0] = *(const v4u*)&Pw[w][lm][kh * 8];
    pf0.u[1] = *(const v4u*)&Pw[w][lm][16 + kh * 8];
    pf1.u[0] = *(const v4u*)&Pw[w][lm][32 + kh * 8];
    pf1.u[1] = *(const v4u*)&Pw[w][lm][48 + kh * 8];

    // O = alpha*O + P*V : 16 WMMAs
#pragma unroll
    for (int d = 0; d < 8; ++d) {
      FragAB vf0, vf1;
      vf0.u[0] = *(const v4u*)&Vs[d * 16 + lm][kh * 8];
      vf0.u[1] = *(const v4u*)&Vs[d * 16 + lm][16 + kh * 8];
      vf1.u[0] = *(const v4u*)&Vs[d * 16 + lm][32 + kh * 8];
      vf1.u[1] = *(const v4u*)&Vs[d * 16 + lm][48 + kh * 8];
      v8f o = O[d];
#pragma unroll
      for (int r = 0; r < 8; ++r) o[r] *= al[r];
      o    = __builtin_amdgcn_wmma_f32_16x16x32_bf16(false, pf0.v, false, vf0.v, (short)0, o, false, false);
      O[d] = __builtin_amdgcn_wmma_f32_16x16x32_bf16(false, pf1.v, false, vf1.v, (short)0, o, false, false);
    }
    __syncthreads();
  }

  // normalize and write bf16 (B,T,D)
#pragma unroll
  for (int r = 0; r < 8; ++r) {
    float  inv    = 1.0f / lrow[r];
    size_t rowoff = ((size_t)(b * SEQ_T + qrow + r + 8 * kh)) * D_MODEL + h * HEAD_DIM;
#pragma unroll
    for (int d = 0; d < 8; ++d)
      Ao[rowoff + d * 16 + lm] = f2bf(O[d][r] * inv);
  }
}

// ---------------------------------------------------------------- launch
extern "C" void kernel_launch(void* const* d_in, const int* in_sizes, int n_in,
                              void* d_out, int out_size, void* d_ws, size_t ws_size,
                              hipStream_t stream) {
  const float* x  = (const float*)d_in[0];
  const float* wq = (const float*)d_in[1];
  const float* wk = (const float*)d_in[2];
  const float* wv = (const float*)d_in[3];
  const float* wo = (const float*)d_in[4];
  const float* bo = (const float*)d_in[5];

  const size_t NX = (size_t)BT * D_MODEL;       // 16.7M elems
  const size_t NW = (size_t)D_MODEL * D_MODEL;  // 4.2M elems

  unsigned short* p   = (unsigned short*)d_ws;  // ~192MB of bf16 scratch
  unsigned short* xb  = p; p += NX;
  unsigned short* wqb = p; p += NW;
  unsigned short* wkb = p; p += NW;
  unsigned short* wvb = p; p += NW;
  unsigned short* wob = p; p += NW;
  unsigned short* qb  = p; p += NX;
  unsigned short* kb  = p; p += NX;
  unsigned short* vtb = p; p += NX;             // (B,H,Dh,T)
  unsigned short* aob = p;                      // (B,T,D)

  const int nx4 = (int)(NX / 4), nw4 = (int)(NW / 4);
  cast_f32_to_bf16_v4<<<(nx4 + 255) / 256, 256, 0, stream>>>(x,  xb,  nx4);
  cast_f32_to_bf16_v4<<<(nw4 + 255) / 256, 256, 0, stream>>>(wq, wqb, nw4);
  cast_f32_to_bf16_v4<<<(nw4 + 255) / 256, 256, 0, stream>>>(wk, wkb, nw4);
  cast_f32_to_bf16_v4<<<(nw4 + 255) / 256, 256, 0, stream>>>(wv, wvb, nw4);
  cast_f32_to_bf16_v4<<<(nw4 + 255) / 256, 256, 0, stream>>>(wo, wob, nw4);

  dim3 gg(D_MODEL / 128, BT / 128);  // (16, 64)
  gemm_bf16_wmma<0><<<gg, 256, 0, stream>>>(xb, wqb, qb,  nullptr, BT, D_MODEL, D_MODEL);
  gemm_bf16_wmma<0><<<gg, 256, 0, stream>>>(xb, wkb, kb,  nullptr, BT, D_MODEL, D_MODEL);
  gemm_bf16_wmma<1><<<gg, 256, 0, stream>>>(xb, wvb, vtb, nullptr, BT, D_MODEL, D_MODEL);

  flash_attn_bf16<<<dim3(BATCH * N_HEADS, SEQ_T / 128), 256, 0, stream>>>(qb, kb, vtb, aob);

  gemm_bf16_wmma<2><<<gg, 256, 0, stream>>>(aob, wob, d_out, bo, BT, D_MODEL, D_MODEL);
}
